// VisionTTTProjector_764504179387
// MI455X (gfx1250) — matrix operand order repaired and checked by reading, not compiled, not run on hardware
//
#include <hip/hip_runtime.h>

// ------------------------------------------------------------------
// VisionTTTProjector for MI455X (gfx1250, wave32, WMMA bf16)
// - Big GEMMs: operands pre-converted to bf16 (A zero-padded row-major,
//   B transposed N x K) -> fragment loads are contiguous dword loads
//   (merge to global_load_b128), no conversion ALU in the hot loop.
// - TTT scan: 48 independent (dir,b,h) states, sequential over 27
//   chunks; all matmuls on v_wmma_f32_16x16x32_bf16, fp32 state and
//   intermediates in L2-resident workspace scratch.
// Workspace use: ~201MB of d_ws.
// ------------------------------------------------------------------

#define B_   4
#define L_   729
#define D_   1152
#define NH_  6
#define HD_  192
#define HD4_ 768
#define K_   27
#define NMB_ 27
#define OUT_ 3584
#define MROWS_ (B_ * L_)        // 2916
#define MPAD_  2928             // 183 * 16
#define BASE_OVER_HD (0.1f / 192.0f)

typedef __attribute__((ext_vector_type(16))) __bf16 v16bf;
typedef __attribute__((ext_vector_type(8)))  float  v8f;

union VB { v16bf v; unsigned u[8]; };

// ---------------- scalar helpers ----------------

__device__ __forceinline__ unsigned short f32_to_bf16u(float f) {
  union { float f32; unsigned u32; } in; in.f32 = f;
  unsigned r = in.u32 + 0x7FFFu + ((in.u32 >> 16) & 1u);   // round-to-nearest-even
  return (unsigned short)(r >> 16);
}

// pack two fp32 -> one dword holding two bf16 (lo, hi)
__device__ __forceinline__ unsigned pack2_bf16(float lo, float hi) {
  union { float f; unsigned u; } a, b;
  a.f = lo; b.f = hi;
  unsigned ua = a.u + 0x7FFFu + ((a.u >> 16) & 1u);
  unsigned ub = b.u + 0x7FFFu + ((b.u >> 16) & 1u);
  return (ua >> 16) | (ub & 0xFFFF0000u);
}

__device__ __forceinline__ float wave_sum(float s) {
  #pragma unroll
  for (int off = 16; off; off >>= 1) s += __shfl_xor(s, off, 32);
  return s;
}

__device__ __forceinline__ float gelu_f(float x) {
  float t = tanhf(0.79788456f * x * (1.0f + 0.044715f * x * x));
  return 0.5f * x * (1.0f + t);
}
__device__ __forceinline__ float gelu_bwd_f(float x) {
  float t = tanhf(0.79788456f * x * (1.0f + 0.044715f * x * x));
  return 0.5f * x * ((1.0f - t * t) * (0.79788456f + 0.1070322243f * x * x)) + 0.5f * (1.0f + t);
}

// release + barrier: cross-CU visibility of global scratch inside the WGP
__device__ __forceinline__ void wg_sync() {
  __threadfence();
  __syncthreads();
}

// ---------------- fragment builders ----------------
// A fragment (16x32, bf16 source, row-major, ld in elements):
// lane m = lane%16; VGPR v holds K pair kb = (lane/16)*16 + 2v
__device__ __forceinline__ v16bf a_frag_bf(const unsigned short* p, int ld, int lane) {
  VB r;
  const unsigned short* q = p + (size_t)(lane & 15) * ld + (lane >> 4) * 16;
  #pragma unroll
  for (int v = 0; v < 8; ++v) {
    int kb = 2 * v;
    r.u[v] = *(const unsigned*)(q + kb);          // 2 contiguous bf16
  }
  return r.v;
}

// B fragment (32x16) from TRANSPOSED bf16 source (N x K, ld = K):
// lane n = lane%16; VGPR v holds K pair k = (lane/16)*16 + 2v  -> contiguous
__device__ __forceinline__ v16bf b_frag_bf_t(const unsigned short* pT, int ld, int lane) {
  VB r;
  const unsigned short* q = pT + (size_t)(lane & 15) * ld + (lane >> 4) * 16;
  #pragma unroll
  for (int v = 0; v < 8; ++v) r.u[v] = *(const unsigned*)(q + 2 * v);
  return r.v;
}

// A fragment from fp32 row-major source (pairs contiguous along K -> float2 loads)
__device__ __forceinline__ v16bf a_frag_f32(const float* src, int ld, int lane) {
  VB r;
  const float* p = src + (size_t)(lane & 15) * ld + (lane >> 4) * 16;
  #pragma unroll
  for (int v = 0; v < 8; ++v) {
    int kb = 2 * v;
    float2 f = *(const float2*)(p + kb);
    r.u[v] = pack2_bf16(f.x, f.y);
  }
  return r.v;
}

// A fragment from a generated expression (non-contiguous source)
template <typename F>
__device__ __forceinline__ v16bf a_frag_gen(F src, int lane) {
  VB r;
  int m = lane & 15, half = lane >> 4;
  #pragma unroll
  for (int v = 0; v < 8; ++v) {
    int kb = 2 * v + half * 16;
    r.u[v] = pack2_bf16(src(m, kb), src(m, kb + 1));
  }
  return r.v;
}

// B fragment from fp32 [K][N] source (pair elements ld apart)
__device__ __forceinline__ v16bf b_frag_f32(const float* src, int ld, int lane) {
  VB r;
  const float* p = src + (lane & 15) + (size_t)(lane >> 4) * 16 * ld;
  #pragma unroll
  for (int v = 0; v < 8; ++v)
    r.u[v] = pack2_bf16(p[(size_t)(2 * v) * ld], p[(size_t)(2 * v + 1) * ld]);
  return r.v;
}

// B fragment from fp32 transposed source: element (k,n) = srcT[n*ld + k] (contiguous)
__device__ __forceinline__ v16bf bt_frag_f32(const float* srcT, int ld, int lane) {
  VB r;
  const float* p = srcT + (size_t)(lane & 15) * ld + (lane >> 4) * 16;
  #pragma unroll
  for (int v = 0; v < 8; ++v) {
    float2 f = *(const float2*)(p + 2 * v);
    r.u[v] = pack2_bf16(f.x, f.y);
  }
  return r.v;
}

// C fragment (fp32): VGPR r holds M = r + 8*(lane/16), N = lane%16
__device__ __forceinline__ v8f c_frag_f32(const float* src, int ld, int lane) {
  v8f c;
  const float* p = src + (lane & 15) + (size_t)(lane >> 4) * 8 * ld;
  #pragma unroll
  for (int r = 0; r < 8; ++r) c[r] = p[(size_t)r * ld];
  return c;
}

__device__ __forceinline__ v8f wmma_bf16(v16bf a, v16bf b, v8f c) {
  return __builtin_amdgcn_wmma_f32_16x16x32_bf16(false, a, false, b, (short)0, c, false, false);
}

// ---------------- bf16 conversion kernels ----------------
// row-major fp32 -> bf16, zero-padding rows [Mreal, Mpad)
__global__ __launch_bounds__(256) void cvt_pad_kernel(const float* __restrict__ src,
                                                      unsigned short* __restrict__ dst,
                                                      int Mreal, int Mpad, int Kd) {
  long i = (long)blockIdx.x * 256 + threadIdx.x;
  if (i >= (long)Mpad * Kd) return;
  int r = (int)(i / Kd);
  dst[i] = (r < Mreal) ? f32_to_bf16u(src[i]) : (unsigned short)0;
}

// fp32 [K][N] -> bf16 transposed [N][K]
__global__ __launch_bounds__(256) void cvt_t_kernel(const float* __restrict__ src,
                                                    unsigned short* __restrict__ dst,
                                                    int Kd, int N) {
  long i = (long)blockIdx.x * 256 + threadIdx.x;
  if (i >= (long)Kd * N) return;
  int k = (int)(i / N), n = (int)(i % N);
  dst[(size_t)n * Kd + k] = f32_to_bf16u(src[i]);
}

// ---------------- GEMM: C[M,N] = A[M,K] @ B[K,N], bf16 operands ----------------
// A: Mpad x Kd bf16 (zero-padded); Bt: N x Kd bf16 (transposed weights)
__global__ __launch_bounds__(256) void gemm_bf16_kernel(const unsigned short* __restrict__ A,
                                                        const unsigned short* __restrict__ Bt,
                                                        float* __restrict__ C,
                                                        int M, int N, int Kd) {
  const int lane = threadIdx.x & 31;
  const int wave = threadIdx.x >> 5;
  const int tilesM  = (M + 15) >> 4;
  const int ngroups = N >> 6;                 // N multiple of 64
  long job = (long)blockIdx.x * 8 + wave;
  if (job >= (long)tilesM * ngroups) return;  // wave-uniform exit
  const int mt = (int)(job / ngroups), ng = (int)(job % ngroups);
  const int m0 = mt * 16, n0 = ng * 64;

  v8f acc[4] = {};
  for (int k0 = 0; k0 < Kd; k0 += 32) {
    if (k0 + 32 < Kd) __builtin_prefetch(A + (size_t)m0 * Kd + k0 + 32, 0, 1);
    v16bf a = a_frag_bf(A + (size_t)m0 * Kd + k0, Kd, lane);
    #pragma unroll
    for (int j = 0; j < 4; ++j) {
      v16bf b = b_frag_bf_t(Bt + (size_t)(n0 + j * 16) * Kd + k0, Kd, lane);
      acc[j] = wmma_bf16(a, b, acc[j]);
    }
  }
  const int nl = lane & 15, half = lane >> 4;
  #pragma unroll
  for (int j = 0; j < 4; ++j) {
    #pragma unroll
    for (int r = 0; r < 8; ++r) {
      int m = m0 + r + half * 8;
      if (m < M) C[(size_t)m * N + n0 + j * 16 + nl] = acc[j][r];
    }
  }
}

// ---------------- rotary (in-place, per head, pos = l % K) ----------------
__global__ __launch_bounds__(256) void rotary_kernel(float* __restrict__ X) {
  long idx = (long)blockIdx.x * 256 + threadIdx.x;
  const long total = (long)MROWS_ * (D_ / 2);
  if (idx >= total) return;
  int tp   = (int)(idx % (D_ / 2));
  long row = idx / (D_ / 2);
  int pos = (int)(row % L_) % K_;
  int u   = tp % (HD_ / 2);
  float inv = powf(10000.0f, -(float)u / (float)(HD_ / 2));
  float ang = (float)pos * inv;
  float c = cosf(ang), s = sinf(ang);
  size_t base = (size_t)row * D_ + 2 * tp;
  float x1 = X[base], x2 = X[base + 1];
  X[base]     = x1 * c - x2 * s;
  X[base + 1] = x1 * s + x2 * c;
}

// ---------------- lr = sigmoid(x . lr_w + lr_b), both directions ----------------
__global__ __launch_bounds__(256) void lr_kernel(const float* __restrict__ x,
                                                 const float* __restrict__ lrw,
                                                 const float* __restrict__ lrb,
                                                 float* __restrict__ LR) {
  const int lane = threadIdx.x & 31, wave = threadIdx.x >> 5;
  long job = (long)blockIdx.x * 8 + wave;
  const long total = 2L * B_ * NH_ * NMB_ * K_;
  if (job >= total) return;
  int k = (int)(job % K_);  long r = job / K_;
  int n = (int)(r % NMB_);  r /= NMB_;
  int h = (int)(r % NH_);   r /= NH_;
  int b = (int)(r % B_);    int dir = (int)(r / B_);
  int l = n * K_ + k; if (dir) l = L_ - 1 - l;
  const float* xr = x + ((size_t)b * L_ + l) * D_;
  const float* w  = lrw + (size_t)h * D_;
  float s = 0.0f;
  for (int c = lane; c < D_; c += 32) s += xr[c] * w[c];
  s = wave_sum(s);
  if (lane == 0) LR[job] = 1.0f / (1.0f + expf(-(s + lrb[h])));
}

// ---------------- TTT scan: one workgroup per (dir, b, h) ----------------
__global__ __launch_bounds__(256) void ttt_kernel(
    const float* __restrict__ XQ, const float* __restrict__ XK, const float* __restrict__ XV,
    const float* __restrict__ W1in, const float* __restrict__ b1in,
    const float* __restrict__ W2in, const float* __restrict__ b2in,
    const float* __restrict__ lnw, const float* __restrict__ lnb,
    const float* __restrict__ LR, const float* __restrict__ tok,
    float* __restrict__ STATE, float* __restrict__ SCR,
    float* __restrict__ accF, float* __restrict__ accB) {
  const int tid  = threadIdx.x;
  const int lane = tid & 31, wave = tid >> 5;
  const int wg  = blockIdx.x;                 // ((dir*B + b)*NH + h)
  const int h   = wg % NH_;
  const int b   = (wg / NH_) % B_;
  const int dir = wg / (NH_ * B_);

  float* W1 = STATE + (size_t)wg * 295872;
  float* b1 = W1 + 147456;
  float* W2 = b1 + 768;
  float* b2 = W2 + 147456;

  float* scr = SCR + (size_t)wg * 160768;
  float* xkb = scr;                 // 32x192
  float* xqb = scr + 6144;          // 32x192
  float* xvb = scr + 12288;         // 32x192
  float* Z1  = scr + 18432;         // 32x768
  float* X2  = scr + 43008;         // 32x768
  float* GZ1 = scr + 67584;         // 32x768
  float* QW1 = scr + 92160;         // 32x768  (xq@W1 + b1)
  float* X2B = scr + 116736;        // 32x768
  float* Z2  = scr + 141312;        // 32x192
  float* GZ2 = scr + 147456;        // 32x192
  float* Z2B = scr + 153600;        // 32x192
  float* ACB = scr + 159744;        // 32x32   (eta-combined attention mask)

  __shared__ float s_ti[32], s_lr[32], s_le[32];
  __shared__ float s_gw[HD_], s_gb[HD_];

  // ---- init state + zero scratch (pad rows must be zero for WMMA padding) ----
  for (int e = tid; e < 147456; e += 256) W1[e] = W1in[(size_t)h * 147456 + e];
  for (int e = tid; e < 147456; e += 256) W2[e] = W2in[(size_t)h * 147456 + e];
  for (int e = tid; e < 768; e += 256) b1[e] = b1in[h * 768 + e];
  if (tid < 192) b2[tid] = b2in[h * 192 + tid];
  for (int e = tid; e < 160768; e += 256) scr[e] = 0.0f;
  if (tid < 32) s_ti[tid] = (tid < K_) ? fmaxf(1.0f / (float)(tid + 1) + tok[tid], 0.0f) : 0.0f;
  if (tid < HD_) { s_gw[tid] = lnw[h * HD_ + tid]; s_gb[tid] = lnb[h * HD_ + tid]; }
  wg_sync();

  for (int n = 0; n < NMB_; ++n) {
    // ---- load chunk (rows reversed for backward direction) ----
    for (int e = tid; e < K_ * HD_; e += 256) {
      int i = e / HD_, c = e % HD_;
      int l = n * K_ + i; int sr = dir ? (L_ - 1 - l) : l;
      size_t src = ((size_t)b * L_ + sr) * D_ + h * HD_ + c;
      xkb[i * HD_ + c] = XK[src];
      xqb[i * HD_ + c] = XQ[src];
      xvb[i * HD_ + c] = XV[src];
    }
    if (tid < 32) {
      float lr = (tid < K_) ? LR[((size_t)wg * NMB_ + n) * K_ + tid] : 0.0f;
      s_lr[tid] = lr;
      s_le[tid] = s_ti[K_ - 1] * BASE_OVER_HD * lr;
    }
    wg_sync();

    // ---- Stage B: Z1 = xk@W1 + b1 ; QW1 = xq@W1 + b1 (192 tiles) ----
    for (int t = wave; t < 192; t += 8) {
      int ob = t / 96, r = t % 96;
      int mt = r / 48, nt = r % 48;
      const float* Abuf = ob ? xqb : xkb;
      int m0 = mt * 16, n0 = nt * 16;
      v8f acc = {};
      for (int k0 = 0; k0 < HD_; k0 += 32) {
        v16bf a  = a_frag_f32(Abuf + m0 * HD_ + k0, HD_, lane);
        v16bf bb = b_frag_f32(W1 + (size_t)k0 * HD4_ + n0, HD4_, lane);
        acc = wmma_bf16(a, bb, acc);
      }
      float* Obuf = ob ? QW1 : Z1;
      int nl = lane & 15, half = lane >> 4;
      #pragma unroll
      for (int rr = 0; rr < 8; ++rr) {
        int m = m0 + rr + half * 8;
        if (m < K_) Obuf[m * HD4_ + n0 + nl] = acc[rr] + b1[n0 + nl];
      }
    }
    wg_sync();

    // ---- Stage C: X2 = gelu(Z1) ----
    for (int e = tid; e < 32 * HD4_; e += 256) X2[e] = gelu_f(Z1[e]);
    wg_sync();

    // ---- Stage D: Z2 = X2@W2 + b2 (24 tiles) ----
    for (int t = wave; t < 24; t += 8) {
      int mt = t / 12, nt = t % 12;
      int m0 = mt * 16, n0 = nt * 16;
      v8f acc = {};
      for (int k0 = 0; k0 < HD4_; k0 += 32) {
        v16bf a  = a_frag_f32(X2 + m0 * HD4_ + k0, HD4_, lane);
        v16bf bb = b_frag_f32(W2 + (size_t)k0 * HD_ + n0, HD_, lane);
        acc = wmma_bf16(a, bb, acc);
      }
      int nl = lane & 15, half = lane >> 4;
      #pragma unroll
      for (int rr = 0; rr < 8; ++rr) {
        int m = m0 + rr + half * 8;
        if (m < K_) Z2[m * HD_ + n0 + nl] = acc[rr] + b2[n0 + nl];
      }
    }
    wg_sync();

    // ---- Stage E: gZ2 = ln_fused_l2_bwd(Z2, xv - xk) (one row per wave) ----
    for (int r = wave; r < K_; r += 8) {
      float z[6], xh[6], gx[6];
      float s0 = 0.0f, s1 = 0.0f;
      #pragma unroll
      for (int j = 0; j < 6; ++j) { z[j] = Z2[r * HD_ + lane + 32 * j]; s0 += z[j]; s1 += z[j] * z[j]; }
      s0 = wave_sum(s0); s1 = wave_sum(s1);
      float mu = s0 * (1.0f / HD_), var = s1 * (1.0f / HD_) - mu * mu;
      float rstd = rsqrtf(var + 1e-6f);
      float sg = 0.0f, sgx = 0.0f;
      #pragma unroll
      for (int j = 0; j < 6; ++j) {
        int c = lane + 32 * j;
        xh[j] = (z[j] - mu) * rstd;
        float tgt = xvb[r * HD_ + c] - xkb[r * HD_ + c];
        gx[j] = (s_gw[c] * xh[j] + s_gb[c] - tgt) * s_gw[c];
        sg += gx[j]; sgx += gx[j] * xh[j];
      }
      sg = wave_sum(sg); sgx = wave_sum(sgx);
      #pragma unroll
      for (int j = 0; j < 6; ++j)
        GZ2[r * HD_ + lane + 32 * j] =
            ((float)HD_ * gx[j] - sg - xh[j] * sgx) * (rstd * (1.0f / HD_));
    }
    wg_sync();

    // ---- Stage F: gZ1 = (gZ2 @ W2^T) * gelu_bwd(Z1) (96 tiles) ----
    for (int t = wave; t < 96; t += 8) {
      int mt = t / 48, nt = t % 48;
      int m0 = mt * 16, n0 = nt * 16;
      v8f acc = {};
      for (int k0 = 0; k0 < HD_; k0 += 32) {
        v16bf a  = a_frag_f32(GZ2 + m0 * HD_ + k0, HD_, lane);
        v16bf bb = bt_frag_f32(W2 + (size_t)n0 * HD_ + k0, HD_, lane);  // W2^T
        acc = wmma_bf16(a, bb, acc);
      }
      int nl = lane & 15, half = lane >> 4;
      #pragma unroll
      for (int rr = 0; rr < 8; ++rr) {
        int m = m0 + rr + half * 8;
        GZ1[m * HD4_ + n0 + nl] = acc[rr] * gelu_bwd_f(Z1[m * HD4_ + n0 + nl]);
      }
    }
    // ---- Stage G: Acomb = eta .* ((xq@xk^T)*tril + tril) (4 tiles) ----
    for (int t = wave; t < 4; t += 8) {
      int mt = t >> 1, nt = t & 1;
      int m0 = mt * 16, n0 = nt * 16;
      v8f acc = {};
      for (int k0 = 0; k0 < HD_; k0 += 32) {
        v16bf a  = a_frag_f32(xqb + m0 * HD_ + k0, HD_, lane);
        v16bf bb = bt_frag_f32(xkb + n0 * HD_ + k0, HD_, lane);         // xk^T
        acc = wmma_bf16(a, bb, acc);
      }
      int nl = lane & 15, half = lane >> 4;
      #pragma unroll
      for (int rr = 0; rr < 8; ++rr) {
        int i = m0 + rr + half * 8, j = n0 + nl;
        float v = 0.0f;
        if (i < K_ && j < K_ && i >= j)
          v = s_ti[i] * BASE_OVER_HD * s_lr[j] * (acc[rr] + 1.0f);
        ACB[i * 32 + j] = v;
      }
    }
    wg_sync();

    // ---- Stage H: X2B = gelu(QW1 - Acomb@gZ1) (96 tiles, 1 k-step) ----
    for (int t = wave; t < 96; t += 8) {
      int mt = t / 48, nt = t % 48;
      int m0 = mt * 16, n0 = nt * 16;
      v16bf a  = a_frag_f32(ACB + m0 * 32, 32, lane);
      v16bf bb = b_frag_f32(GZ1 + n0, HD4_, lane);
      v8f acc = {};
      acc = wmma_bf16(a, bb, acc);
      int nl = lane & 15, half = lane >> 4;
      #pragma unroll
      for (int rr = 0; rr < 8; ++rr) {
        int m = m0 + rr + half * 8;
        if (m < K_) X2B[m * HD4_ + n0 + nl] = gelu_f(QW1[m * HD4_ + n0 + nl] - acc[rr]);
      }
    }
    wg_sync();

    // ---- Stage I: Bcomb = eta .* ((X2B@X2^T)*tril + tril) (reuse ACB) ----
    for (int t = wave; t < 4; t += 8) {
      int mt = t >> 1, nt = t & 1;
      int m0 = mt * 16, n0 = nt * 16;
      v8f acc = {};
      for (int k0 = 0; k0 < HD4_; k0 += 32) {
        v16bf a  = a_frag_f32(X2B + m0 * HD4_ + k0, HD4_, lane);
        v16bf bb = bt_frag_f32(X2 + (size_t)n0 * HD4_ + k0, HD4_, lane); // X2^T
        acc = wmma_bf16(a, bb, acc);
      }
      int nl = lane & 15, half = lane >> 4;
      #pragma unroll
      for (int rr = 0; rr < 8; ++rr) {
        int i = m0 + rr + half * 8, j = n0 + nl;
        float v = 0.0f;
        if (i < K_ && j < K_ && i >= j)
          v = s_ti[i] * BASE_OVER_HD * s_lr[j] * (acc[rr] + 1.0f);
        ACB[i * 32 + j] = v;
      }
    }
    wg_sync();

    // ---- Stage J: Z2B = X2B@W2 - Bcomb@gZ2 + b2 (24 tiles) ----
    for (int t = wave; t < 24; t += 8) {
      int mt = t / 12, nt = t % 12;
      int m0 = mt * 16, n0 = nt * 16;
      v8f acc = {};
      for (int k0 = 0; k0 < HD4_; k0 += 32) {
        v16bf a  = a_frag_f32(X2B + m0 * HD4_ + k0, HD4_, lane);
        v16bf bb = b_frag_f32(W2 + (size_t)k0 * HD_ + n0, HD_, lane);
        acc = wmma_bf16(a, bb, acc);
      }
      v16bf a2  = a_frag_f32(ACB + m0 * 32, 32, lane);
      v16bf bb2 = b_frag_f32(GZ2 + n0, HD_, lane);
      v8f acc2 = {};
      acc2 = wmma_bf16(a2, bb2, acc2);
      int nl = lane & 15, half = lane >> 4;
      #pragma unroll
      for (int rr = 0; rr < 8; ++rr) {
        int m = m0 + rr + half * 8;
        if (m < K_) Z2B[m * HD_ + n0 + nl] = acc[rr] - acc2[rr] + b2[n0 + nl];
      }
    }
    wg_sync();

    // ---- Stage K: rank-27 state updates (C = live weights, A pre-negated) ----
    for (int t = wave; t < 1152; t += 8) {
      if (t < 576) {            // W1 -= (le.*xk)^T @ gZ1
        int mt = t / 48, nt = t % 48;
        int m0 = mt * 16, n0 = nt * 16;
        v16bf a  = a_frag_gen([&](int m, int k) { return -s_le[k] * xkb[k * HD_ + m0 + m]; }, lane);
        v16bf bb = b_frag_f32(GZ1 + n0, HD4_, lane);
        v8f c = c_frag_f32(W1 + (size_t)m0 * HD4_ + n0, HD4_, lane);
        v8f d = wmma_bf16(a, bb, c);
        int nl = lane & 15, half = lane >> 4;
        #pragma unroll
        for (int rr = 0; rr < 8; ++rr)
          W1[(size_t)(m0 + rr + half * 8) * HD4_ + n0 + nl] = d[rr];
      } else {                  // W2 -= (le.*X2)^T @ gZ2
        int tt = t - 576;
        int mt = tt / 12, nt = tt % 12;
        int m0 = mt * 16, n0 = nt * 16;
        v16bf a  = a_frag_gen([&](int m, int k) { return -s_le[k] * X2[k * HD4_ + m0 + m]; }, lane);
        v16bf bb = b_frag_f32(GZ2 + n0, HD_, lane);
        v8f c = c_frag_f32(W2 + (size_t)m0 * HD_ + n0, HD_, lane);
        v8f d = wmma_bf16(a, bb, c);
        int nl = lane & 15, half = lane >> 4;
        #pragma unroll
        for (int rr = 0; rr < 8; ++rr)
          W2[(size_t)(m0 + rr + half * 8) * HD_ + n0 + nl] = d[rr];
      }
    }
    // bias updates
    for (int c = tid; c < HD4_; c += 256) {
      float s = 0.0f;
      for (int i = 0; i < K_; ++i) s += s_le[i] * GZ1[i * HD4_ + c];
      b1[c] -= s;
    }
    if (tid < HD_) {
      float s = 0.0f;
      for (int i = 0; i < K_; ++i) s += s_le[i] * GZ2[i * HD_ + tid];
      b2[tid] -= s;
    }
    // ---- Stage L: out = xq + ln_fwd(Z2B) ----
    for (int r = wave; r < K_; r += 8) {
      float z[6]; float s0 = 0.0f, s1 = 0.0f;
      #pragma unroll
      for (int j = 0; j < 6; ++j) { z[j] = Z2B[r * HD_ + lane + 32 * j]; s0 += z[j]; s1 += z[j] * z[j]; }
      s0 = wave_sum(s0); s1 = wave_sum(s1);
      float mu = s0 * (1.0f / HD_), var = s1 * (1.0f / HD_) - mu * mu;
      float rstd = rsqrtf(var + 1e-6f);
      int l = n * K_ + r; int grow = dir ? (L_ - 1 - l) : l;
      float* dst = (dir ? accB : accF) + ((size_t)b * L_ + grow) * D_ + h * HD_;
      #pragma unroll
      for (int j = 0; j < 6; ++j) {
        int c = lane + 32 * j;
        dst[c] = xqb[r * HD_ + c] + s_gw[c] * ((z[j] - mu) * rstd) + s_gb[c];
      }
    }
    wg_sync();
  }
}

// ---------------- gate * (out_f + out_b), post-LN (eps 1e-5), in-place into G ----------------
__global__ __launch_bounds__(256) void gate_ln_kernel(const float* __restrict__ G,
                                                      const float* __restrict__ aF,
                                                      const float* __restrict__ aB,
                                                      const float* __restrict__ pw,
                                                      const float* __restrict__ pb,
                                                      float* __restrict__ NRM) {
  __shared__ float r1[8], r2[8];
  const size_t row = blockIdx.x;
  const int t = threadIdx.x, lane = t & 31, wave = t >> 5;
  float vals[5];
  float s1 = 0.0f, s2 = 0.0f;
  int cnt = 0;
  for (int c = t; c < D_; c += 256) {
    float v = gelu_f(G[row * D_ + c]) * (aF[row * D_ + c] + aB[row * D_ + c]);
    vals[cnt++] = v; s1 += v; s2 += v * v;
  }
  s1 = wave_sum(s1); s2 = wave_sum(s2);
  if (lane == 0) { r1[wave] = s1; r2[wave] = s2; }
  __syncthreads();
  float S1 = 0.0f, S2 = 0.0f;
  #pragma unroll
  for (int i = 0; i < 8; ++i) { S1 += r1[i]; S2 += r2[i]; }
  float mu = S1 * (1.0f / D_), var = S2 * (1.0f / D_) - mu * mu;
  float rstd = rsqrtf(var + 1e-5f);
  cnt = 0;
  for (int c = t; c < D_; c += 256)
    NRM[row * D_ + c] = (vals[cnt++] - mu) * rstd * pw[c] + pb[c];
}

// ---------------- host launcher ----------------
extern "C" void kernel_launch(void* const* d_in, const int* in_sizes, int n_in,
                              void* d_out, int out_size, void* d_ws, size_t ws_size,
                              hipStream_t stream) {
  (void)in_sizes; (void)n_in; (void)out_size; (void)ws_size;
  const float* x    = (const float*)d_in[0];
  const float* wq   = (const float*)d_in[1];
  const float* wk   = (const float*)d_in[2];
  const float* wv   = (const float*)d_in[3];
  const float* wo   = (const float*)d_in[4];
  const float* wg   = (const float*)d_in[5];
  const float* W1in = (const float*)d_in[6];
  const float* b1in = (const float*)d_in[7];
  const float* W2in = (const float*)d_in[8];
  const float* b2in = (const float*)d_in[9];
  const float* lnw  = (const float*)d_in[10];
  const float* lnb  = (const float*)d_in[11];
  const float* pw   = (const float*)d_in[12];
  const float* pb   = (const float*)d_in[13];
  const float* lrw  = (const float*)d_in[14];
  const float* lrb  = (const float*)d_in[15];
  const float* tok  = (const float*)d_in[16];

  float* ws = (float*)d_ws;
  const size_t S = (size_t)MROWS_ * D_;       // 3,359,232 floats
  float* XQ = ws;
  float* XK = ws + S;
  float* XV = ws + 2 * S;
  float* G  = ws + 3 * S;                     // gate pre-act; reused as LN output
  float* aF = ws + 4 * S;
  float* aB = ws + 5 * S;
  float* LR = ws + 6 * S;                     // 34,992 floats (reserve 35,072)
  float* ST = LR + 35072;                     // 48 * 295,872 floats
  float* SC = ST + (size_t)48 * 295872;       // 48 * 160,768 floats
  float* FEND = SC + (size_t)48 * 160768;

  // bf16 region (ushorts), overlayed after the fp32 region
  unsigned short* us  = (unsigned short*)FEND;
  const size_t SP = (size_t)MPAD_ * D_;       // 3,373,056
  const size_t WSZ = (size_t)D_ * D_;         // 1,327,104
  unsigned short* XB  = us;                   // x, bf16, padded rows
  unsigned short* GB  = XB + SP;              // gated output, bf16, padded rows
  unsigned short* WQT = GB + SP;              // transposed bf16 weights (N x K)
  unsigned short* WKT = WQT + WSZ;
  unsigned short* WVT = WKT + WSZ;
  unsigned short* WGT = WVT + WSZ;
  unsigned short* WOT = WGT + WSZ;            // OUT_ x D_

  const int M = MROWS_;
  const int tilesM = MPAD_ / 16;              // 183

  // ---- operand conversion (bf16 / transposed) ----
  {
    long n = (long)MPAD_ * D_;
    cvt_pad_kernel<<<(int)((n + 255) / 256), 256, 0, stream>>>(x, XB, M, MPAD_, D_);
    long w = (long)D_ * D_;
    cvt_t_kernel<<<(int)((w + 255) / 256), 256, 0, stream>>>(wq, WQT, D_, D_);
    cvt_t_kernel<<<(int)((w + 255) / 256), 256, 0, stream>>>(wk, WKT, D_, D_);
    cvt_t_kernel<<<(int)((w + 255) / 256), 256, 0, stream>>>(wv, WVT, D_, D_);
    cvt_t_kernel<<<(int)((w + 255) / 256), 256, 0, stream>>>(wg, WGT, D_, D_);
    long wo_n = (long)D_ * OUT_;
    cvt_t_kernel<<<(int)((wo_n + 255) / 256), 256, 0, stream>>>(wo, WOT, D_, OUT_);
  }

  // ---- projections ----
  const int blocksP = (tilesM * (D_ / 64) + 7) / 8;
  gemm_bf16_kernel<<<blocksP, 256, 0, stream>>>(XB, WQT, XQ, M, D_, D_);
  gemm_bf16_kernel<<<blocksP, 256, 0, stream>>>(XB, WKT, XK, M, D_, D_);
  gemm_bf16_kernel<<<blocksP, 256, 0, stream>>>(XB, WVT, XV, M, D_, D_);
  gemm_bf16_kernel<<<blocksP, 256, 0, stream>>>(XB, WGT, G,  M, D_, D_);

  const long rt = (long)M * (D_ / 2);
  rotary_kernel<<<(int)((rt + 255) / 256), 256, 0, stream>>>(XQ);
  rotary_kernel<<<(int)((rt + 255) / 256), 256, 0, stream>>>(XK);

  const int lrJobs = 2 * B_ * NH_ * NMB_ * K_;
  lr_kernel<<<(lrJobs + 7) / 8, 256, 0, stream>>>(x, lrw, lrb, LR);

  ttt_kernel<<<2 * B_ * NH_, 256, 0, stream>>>(XQ, XK, XV, W1in, b1in, W2in, b2in,
                                               lnw, lnb, LR, tok, ST, SC, aF, aB);

  gate_ln_kernel<<<M, 256, 0, stream>>>(G, aF, aB, pw, pb, G);

  // ---- output GEMM: (gated,normed) @ wo ----
  {
    long n = (long)MPAD_ * D_;
    cvt_pad_kernel<<<(int)((n + 255) / 256), 256, 0, stream>>>(G, GB, M, MPAD_, D_);
  }
  const int blocksO = (tilesM * (OUT_ / 64) + 7) / 8;
  gemm_bf16_kernel<<<blocksO, 256, 0, stream>>>(GB, WOT, (float*)d_out, M, OUT_, D_);
}
